// PVADiffusion_13400297963868
// MI455X (gfx1250) — matrix operand order repaired and checked by reading, not compiled, not run on hardware
//
#include <hip/hip_runtime.h>

// ---- problem dims (fixed by the reference) ----
#define B_    16
#define S_    4096
#define QD_   640      // query model dim
#define CD_   768      // context (enc) dim
#define ID_   640      // inner dim = H*D
#define H_    8
#define D_    80       // head dim (padded to 96 in K-chunks)
#define L_    87       // text(77) + visual(10) keys
#define LP_   96       // keys padded to 6 x 16
#define TEXT_ 77

typedef __attribute__((ext_vector_type(16))) __bf16 v16bf;
typedef __attribute__((ext_vector_type(8)))  float  v8f;

union Frag { v16bf v; unsigned int w[8]; unsigned short us[16]; };

// hardware f32 -> bf16 (RNE) via native cast; returned as raw bits
__device__ __forceinline__ unsigned short f2bf(float f) {
  union { __bf16 h; unsigned short u; } c;
  c.h = (__bf16)f;
  return c.u;
}

// ---- A fragment (16x32 bf16, ISA 7.12.2 16-bit A layout) from f32 row-major ----
// Row index is CLAMPED (not predicated) so no exec-mask branches are generated;
// rows >= rowvalid produce duplicate (finite) data that callers mask at store time.
__device__ __forceinline__ v16bf load_a_f32(const float* base, int ldk, int rowvalid) {
  int lane = threadIdx.x & 31;
  int row  = lane & 15;
  int rc   = (row < rowvalid) ? row : (rowvalid - 1);
  int kb   = (lane & 16) ? 8 : 0;
  const float* r = base + (size_t)rc * ldk;
  Frag f;
#pragma unroll
  for (int i = 0; i < 8; ++i) {
    int k = kb + ((i < 4) ? (i << 1) : (16 + ((i - 4) << 1)));
    f.us[2 * i]     = f2bf(r[k]);
    f.us[2 * i + 1] = f2bf(r[k + 1]);
  }
  return f.v;
}

// ---- A fragment from bf16 (ushort) row-major storage; k offsets all even -> dword loads ----
__device__ __forceinline__ v16bf load_a_bf16(const unsigned short* base, int ldk, int kvalid) {
  int lane = threadIdx.x & 31;
  int row  = lane & 15;
  int kb   = (lane & 16) ? 8 : 0;
  const unsigned int* r = (const unsigned int*)(base + (size_t)row * ldk);
  Frag f;
#pragma unroll
  for (int i = 0; i < 8; ++i) {
    int k = kb + ((i < 4) ? (i << 1) : (16 + ((i - 4) << 1)));
    f.w[i] = (k < kvalid) ? r[k >> 1] : 0u;
  }
  return f.v;
}

// ---- B fragment (32x16 bf16). Storage N-major rows, K contiguous (B^T row-major). ----
__device__ __forceinline__ v16bf load_b_bf16(const unsigned short* base, int ldk, int kvalid) {
  int lane = threadIdx.x & 31;
  int col  = lane & 15;
  int kb   = (lane & 16) ? 8 : 0;
  const unsigned int* r = (const unsigned int*)(base + (size_t)col * ldk);
  Frag f;
#pragma unroll
  for (int i = 0; i < 8; ++i) {
    int k = kb + ((i < 4) ? (i << 1) : (16 + ((i - 4) << 1)));
    f.w[i] = (k < kvalid) ? r[k >> 1] : 0u;
  }
  return f.v;
}

#define WMMA_BF16(A, B, C) \
  __builtin_amdgcn_wmma_f32_16x16x32_bf16(false, (A), false, (B), (short)0, (C), false, false)

#define ASYNC_LDS_B128(ldsoff, gptr) \
  asm volatile("global_load_async_to_lds_b128 %0, %1, off" :: "v"(ldsoff), "v"(gptr) : "memory")
#define WAIT_ASYNC() asm volatile("s_wait_asynccnt 0x0" ::: "memory")

// ---- convert+transpose a weight: W (K x N, f32 row-major) -> Wt (N x K, bf16) ----
__global__ void k_wt(const float* __restrict__ W, unsigned short* __restrict__ Wt, int K, int N) {
  int i = blockIdx.x * 256 + threadIdx.x;
  if (i >= K * N) return;
  int n = i / K, k = i - n * K;
  Wt[i] = f2bf(W[(size_t)k * N + n]);
}

// ---- bulk f32 -> bf16 conversion (x), vectorized 4 elems/thread ----
__global__ void k_x2bf(const float* __restrict__ x, unsigned short* __restrict__ xbf,
                       long long nelem) {
  long long i = ((long long)blockIdx.x * 256 + threadIdx.x) * 4;
  if (i >= nelem) return;
  float4 v = *(const float4*)(x + i);
  union { unsigned short us[4]; uint2 u2; } o;
  o.us[0] = f2bf(v.x); o.us[1] = f2bf(v.y); o.us[2] = f2bf(v.z); o.us[3] = f2bf(v.w);
  *(uint2*)(xbf + i) = o.u2;
}

// ---- generic big GEMM: C[M x N] = A[M x K](bf16) @ Bt[N x K](bf16)^T (+bias) ----
// block = 256 thr (8 waves) -> 128(M) x 128(N); wave -> 16 x 128 (8 accs).
// B panels (128x32 bf16 = 8KB) double-buffered in LDS via async copies; A fragment
// software-pipelined one K-chunk ahead so global loads overlap the 8 WMMAs.
template <bool OUT_F32>
__global__ void __launch_bounds__(256) k_gemm(const unsigned short* __restrict__ A,
                                              const unsigned short* __restrict__ Bt,
                                              const float* __restrict__ bias,
                                              unsigned short* __restrict__ Cbf,
                                              float* __restrict__ Cf,
                                              int K, int ldc) {
  __shared__ __align__(16) unsigned short bpan[2][128 * 32];
  int w    = threadIdx.x >> 5;
  int lane = threadIdx.x & 31;
  int col0 = blockIdx.x * 128;
  int row0 = blockIdx.y * 128 + w * 16;
  int nk   = K >> 5;

  auto stage = [&](int kc, int bf) {
    int t = threadIdx.x;
#pragma unroll
    for (int p = 0; p < 2; ++p) {
      int r   = (t >> 2) + p * 64;           // panel row 0..127
      int sub = (t & 3) * 8;                 // 16B sub-chunk within the 64B row
      const unsigned short* g = Bt + (size_t)(col0 + r) * K + kc * 32 + sub;
      unsigned int ldsoff = (unsigned int)(unsigned long long)&bpan[bf][r * 32 + sub];
      ASYNC_LDS_B128(ldsoff, g);
    }
  };

  v8f acc[8] = {};
  const unsigned short* arow = A + (size_t)row0 * K;

  v16bf a_cur = load_a_bf16(arow, K, 32);
  stage(0, 0);
  for (int kc = 0; kc < nk; ++kc) {
    int cur = kc & 1;
    WAIT_ASYNC();
    __syncthreads();                          // panel[cur] complete + visible to all waves
    v16bf a_next = a_cur;
    if (kc + 1 < nk) {
      stage(kc + 1, cur ^ 1);                 // overlap next B panel with WMMA
      a_next = load_a_bf16(arow + (kc + 1) * 32, K, 32);  // pipeline next A fragment
    }
#pragma unroll
    for (int j = 0; j < 8; ++j) {
      v16bf b = load_b_bf16(&bpan[cur][(16 * j) * 32], 32, 32);
      acc[j] = WMMA_BF16(a_cur, b, acc[j]);
    }
    a_cur = a_next;
    __syncthreads();                          // all waves done reading before overwrite
  }

  int n = lane & 15, mb = (lane & 16) ? 8 : 0;
#pragma unroll
  for (int j = 0; j < 8; ++j) {
    int ng = col0 + 16 * j + n;
    if constexpr (OUT_F32) {
      float bv = bias[ng];
#pragma unroll
      for (int vv = 0; vv < 8; ++vv)
        Cf[(size_t)(row0 + mb + vv) * ldc + ng] = acc[j][vv] + bv;
    } else {
#pragma unroll
      for (int vv = 0; vv < 8; ++vv)
        Cbf[(size_t)(row0 + mb + vv) * ldc + ng] = f2bf(acc[j][vv]);
    }
  }
}

// ---- enc projections -> K (key-major, d contiguous) and V^T (d-major, key contiguous, L->96)
__global__ void __launch_bounds__(32) k_proj_kv(const float* __restrict__ enc,
                                                const unsigned short* __restrict__ WkT,
                                                const unsigned short* __restrict__ WvT,
                                                const unsigned short* __restrict__ WvkT,
                                                const unsigned short* __restrict__ WvvT,
                                                unsigned short* __restrict__ k_all,
                                                unsigned short* __restrict__ v_t) {
  int ntile = blockIdx.x, ltile = blockIdx.y, b = blockIdx.z;
  int lane = threadIdx.x & 31;
  int row0 = ltile * 16, col0 = ntile * 16;
  int rowvalid = L_ - row0;
  if (rowvalid > 16) rowvalid = 16;           // >= 7 for every launched tile
  v8f akt = {}, akv = {}, avt = {}, avv = {};
  const float* erow = enc + ((size_t)b * L_ + row0) * CD_;
  for (int kc = 0; kc < CD_ / 32; ++kc) {
    v16bf a  = load_a_f32(erow + kc * 32, CD_, rowvalid);
    v16bf b0 = load_b_bf16(WkT  + (size_t)col0 * CD_ + kc * 32, CD_, 32);
    akt = WMMA_BF16(a, b0, akt);
    v16bf b1 = load_b_bf16(WvkT + (size_t)col0 * CD_ + kc * 32, CD_, 32);
    akv = WMMA_BF16(a, b1, akv);
    v16bf b2 = load_b_bf16(WvT  + (size_t)col0 * CD_ + kc * 32, CD_, 32);
    avt = WMMA_BF16(a, b2, avt);
    v16bf b3 = load_b_bf16(WvvT + (size_t)col0 * CD_ + kc * 32, CD_, 32);
    avv = WMMA_BF16(a, b3, avv);
  }
  int n = lane & 15, mb = (lane & 16) ? 8 : 0;
  int nglob = col0 + n;
  int h = nglob / D_, d = nglob - h * D_;
#pragma unroll
  for (int vv = 0; vv < 8; ++vv) {
    int l = row0 + mb + vv;                 // < 96 always
    float kval = 0.f, vval = 0.f;
    if (l < L_) {
      kval = (l < TEXT_) ? akt[vv] : akv[vv];
      vval = (l < TEXT_) ? avt[vv] : avv[vv];
    }
    k_all[((size_t)b * LP_ + l) * ID_ + nglob]     = f2bf(kval);  // zero pad rows 87..95
    v_t[(((size_t)b * H_ + h) * D_ + d) * LP_ + l] = f2bf(vval);  // zero pad keys 87..95
  }
}

// ---- attention: per (b,h), 8 waves x 16 query rows. K and V tiles (15KB each) are
// async-staged into LDS once per block (all 8 waves share them). WMMA scores ->
// wave32 softmax (shfl_xor over 16-lane halves) -> LDS transpose -> WMMA P@V. ----
__global__ void __launch_bounds__(256) k_attn(const unsigned short* __restrict__ q_all,
                                              const unsigned short* __restrict__ k_all,
                                              const unsigned short* __restrict__ v_t,
                                              unsigned short* __restrict__ attn_bf) {
  __shared__ __align__(16) unsigned short kt[LP_ * D_];      // 96 keys x 80 d
  __shared__ __align__(16) unsigned short vt[D_ * LP_];      // 80 d x 96 keys
  __shared__ __align__(16) unsigned short plds[8][16 * LP_]; // probs, per wave
  int w = threadIdx.x >> 5, lane = threadIdx.x & 31;
  int tid = threadIdx.x;
  int h = blockIdx.y, b = blockIdx.z;
  int q0 = blockIdx.x * 128 + w * 16;
  size_t rowbase = (size_t)b * S_ + q0;
  const float scale = 0.11180339887498949f;  // 1/sqrt(80)

  // stage K tile: 96 rows x 160B = 960 16B chunks
#pragma unroll
  for (int p = 0; p < 4; ++p) {
    int c = p * 256 + tid;
    if (c < 960) {
      int l = c / 10, sub = c - l * 10;
      const unsigned short* g = k_all + ((size_t)b * LP_ + l) * ID_ + h * D_ + sub * 8;
      unsigned int ldsoff = (unsigned int)(unsigned long long)&kt[l * D_ + sub * 8];
      ASYNC_LDS_B128(ldsoff, g);
    }
  }
  // stage V tile: 80 rows x 192B = 960 16B chunks
#pragma unroll
  for (int p = 0; p < 4; ++p) {
    int c = p * 256 + tid;
    if (c < 960) {
      int dd = c / 12, sub = c - dd * 12;
      const unsigned short* g = v_t + (((size_t)b * H_ + h) * D_ + dd) * LP_ + sub * 8;
      unsigned int ldsoff = (unsigned int)(unsigned long long)&vt[dd * LP_ + sub * 8];
      ASYNC_LDS_B128(ldsoff, g);
    }
  }

  // overlap: Q fragments from global while the async copies fly
  v16bf At[3], Av[3];
#pragma unroll
  for (int kc = 0; kc < 3; ++kc) {
    int kv = D_ - kc * 32; if (kv > 32) kv = 32;          // 32,32,16
    At[kc] = load_a_bf16(q_all + rowbase * (2 * ID_) + h * D_ + kc * 32, 2 * ID_, kv);
    Av[kc] = load_a_bf16(q_all + rowbase * (2 * ID_) + ID_ + h * D_ + kc * 32, 2 * ID_, kv);
  }

  WAIT_ASYNC();
  __syncthreads();                            // K/V tiles resident in LDS

  v8f sc[6];
#pragma unroll
  for (int t = 0; t < 6; ++t) {
    v8f st = {}, sv = {};
    bool needT = (t <= 4);   // tiles touching keys < 77
    bool needV = (t >= 4);   // tiles touching keys >= 77
#pragma unroll
    for (int kc = 0; kc < 3; ++kc) {
      int kv = D_ - kc * 32; if (kv > 32) kv = 32;
      v16bf bk = load_b_bf16(&kt[(t * 16) * D_ + kc * 32], D_, kv);
      if (needT) st = WMMA_BF16(At[kc], bk, st);
      if (needV) sv = WMMA_BF16(Av[kc], bk, sv);
    }
    int key = t * 16 + (lane & 15);          // this lane's key column
#pragma unroll
    for (int vv = 0; vv < 8; ++vv) {
      float val = ((key < TEXT_) ? st[vv] : sv[vv]) * scale;
      if (key >= L_) val = -1e30f;           // mask padded keys
      sc[t][vv] = val;
    }
  }

  float rmax[8], rinv[8];
#pragma unroll
  for (int vv = 0; vv < 8; ++vv) {
    float m = sc[0][vv];
#pragma unroll
    for (int t = 1; t < 6; ++t) m = fmaxf(m, sc[t][vv]);
#pragma unroll
    for (int mk = 1; mk < 16; mk <<= 1) m = fmaxf(m, __shfl_xor(m, mk, 32));
    rmax[vv] = m;
  }
#pragma unroll
  for (int vv = 0; vv < 8; ++vv) {
    float s = 0.f;
#pragma unroll
    for (int t = 0; t < 6; ++t) { float e = __expf(sc[t][vv] - rmax[vv]); sc[t][vv] = e; s += e; }
#pragma unroll
    for (int mk = 1; mk < 16; mk <<= 1) s += __shfl_xor(s, mk, 32);
    rinv[vv] = 1.f / s;
  }

  int n = lane & 15, mb = (lane & 16) ? 8 : 0;
#pragma unroll
  for (int t = 0; t < 6; ++t)
#pragma unroll
    for (int vv = 0; vv < 8; ++vv)
      plds[w][(mb + vv) * LP_ + t * 16 + n] = f2bf(sc[t][vv] * rinv[vv]);
  __syncthreads();

  v16bf P[3];
#pragma unroll
  for (int kc = 0; kc < 3; ++kc) P[kc] = load_a_bf16(&plds[w][0] + kc * 32, LP_, 32);

#pragma unroll
  for (int nt = 0; nt < 5; ++nt) {          // d tiles: 5 x 16 = 80
    v8f acc = {};
#pragma unroll
    for (int kc = 0; kc < 3; ++kc) {
      v16bf bv = load_b_bf16(&vt[(nt * 16) * LP_ + kc * 32], LP_, 32);
      acc = WMMA_BF16(P[kc], bv, acc);
    }
#pragma unroll
    for (int vv = 0; vv < 8; ++vv)
      attn_bf[(rowbase + mb + vv) * ID_ + h * D_ + nt * 16 + n] = f2bf(acc[vv]);
  }
}

extern "C" void kernel_launch(void* const* d_in, const int* in_sizes, int n_in,
                              void* d_out, int out_size, void* d_ws, size_t ws_size,
                              hipStream_t stream) {
  (void)in_sizes; (void)n_in; (void)out_size; (void)ws_size;
  const float* x   = (const float*)d_in[0];
  const float* enc = (const float*)d_in[1];
  const float* Wq  = (const float*)d_in[2];
  const float* Wvq = (const float*)d_in[3];
  const float* Wk  = (const float*)d_in[4];
  const float* Wv  = (const float*)d_in[5];
  const float* Wvk = (const float*)d_in[6];
  const float* Wvv = (const float*)d_in[7];
  const float* Wo  = (const float*)d_in[8];
  const float* bo  = (const float*)d_in[9];
  float* out = (float*)d_out;

  const size_t BS = (size_t)B_ * S_;
  char* ws = (char*)d_ws;
  size_t off = 0;
  auto carve = [&](size_t elems) -> unsigned short* {
    unsigned short* p = (unsigned short*)(ws + off);
    off += ((elems * 2) + 255) & ~(size_t)255;
    return p;
  };
  unsigned short* WcombT = carve((size_t)2 * ID_ * QD_);      // [1280 x 640] Wq^T | Wvq^T
  unsigned short* WkT    = carve((size_t)ID_ * CD_);
  unsigned short* WvT    = carve((size_t)ID_ * CD_);
  unsigned short* WvkT   = carve((size_t)ID_ * CD_);
  unsigned short* WvvT   = carve((size_t)ID_ * CD_);
  unsigned short* WoT    = carve((size_t)QD_ * ID_);
  unsigned short* x_bf   = carve(BS * QD_);                   // [BS x 640] bf16 (fits in L2)
  unsigned short* q_all  = carve(BS * 2 * ID_);               // [BS x 1280] tq|vq bf16
  unsigned short* k_all  = carve((size_t)B_ * LP_ * ID_);     // [B x 96 x 640] key-major
  unsigned short* v_t    = carve((size_t)B_ * H_ * D_ * LP_); // [B x H x 80 x 96] d-major
  unsigned short* attn   = carve(BS * ID_);                   // [BS x 640] bf16

  // 1) weight convert+transpose (bf16, N-major/K-contiguous for B fragments)
  k_wt<<<(QD_ * ID_ + 255) / 256, 256, 0, stream>>>(Wq,  WcombT,                     QD_, ID_);
  k_wt<<<(QD_ * ID_ + 255) / 256, 256, 0, stream>>>(Wvq, WcombT + (size_t)ID_ * QD_, QD_, ID_);
  k_wt<<<(CD_ * ID_ + 255) / 256, 256, 0, stream>>>(Wk,  WkT,  CD_, ID_);
  k_wt<<<(CD_ * ID_ + 255) / 256, 256, 0, stream>>>(Wv,  WvT,  CD_, ID_);
  k_wt<<<(CD_ * ID_ + 255) / 256, 256, 0, stream>>>(Wvk, WvkT, CD_, ID_);
  k_wt<<<(CD_ * ID_ + 255) / 256, 256, 0, stream>>>(Wvv, WvvT, CD_, ID_);
  k_wt<<<(ID_ * QD_ + 255) / 256, 256, 0, stream>>>(Wo,  WoT,  ID_, QD_);

  // 2) x -> bf16 once
  long long nx = (long long)BS * QD_;
  k_x2bf<<<(unsigned)(nx / 4 / 256), 256, 0, stream>>>(x, x_bf, nx);

  // 3) q/vq projection: [BS x 1280] = x_bf @ WcombT^T   (async-LDS staged, bf16 out)
  k_gemm<false><<<dim3((2 * ID_) / 128, BS / 128), 256, 0, stream>>>(
      x_bf, WcombT, nullptr, q_all, nullptr, QD_, 2 * ID_);

  // 4) K/V projections (tiny), writes zero padding for keys 87..95
  k_proj_kv<<<dim3(ID_ / 16, LP_ / 16, B_), 32, 0, stream>>>(enc, WkT, WvT, WvkT, WvvT, k_all, v_t);

  // 5) attention (scores -> softmax -> P@V), K/V async-staged to LDS, all matmuls on WMMA
  k_attn<<<dim3(S_ / 128, H_, B_), 256, 0, stream>>>(q_all, k_all, v_t, attn);

  // 6) output GEMM + bias (async-LDS staged, f32 out)
  k_gemm<true><<<dim3(QD_ / 128, BS / 128), 256, 0, stream>>>(
      attn, WoT, bo, nullptr, out, ID_, QD_);
}
